// MaxPooling_89223650607313
// MI455X (gfx1250) — compile-verified, matrix operand order checked by reading, not run on hardware
//
#include <hip/hip_runtime.h>
#include <math.h>

// Problem constants from the reference: x[B=16, T=4096, D=1024] f32, mask[B,T] i32.
#define BB 16
#define TT 4096
#define DD 1024
#define NSEG 64           // time segments per batch (lengths ~2048 -> ~32 active/b)
#define TSEG (TT / NSEG)  // 64 rows per block

typedef __attribute__((ext_vector_type(8))) int v8i;

// ---------------------------------------------------------------------------
// Kernel 1: lengths[b] = sum_t (1 - mask[b][t]) via V_WMMA_I32_16X16X64_IU8.
// One wave. A (16x64 u8) = packed (1-mask) tile, B = all-ones bytes, so
// D[m][n] = sum_k A[m][k]; K-layout permutation is irrelevant with ones-B.
// Lanes 0-15 supply K 0..31 of row m=lane, lanes 16-31 supply K 32..63.
// ---------------------------------------------------------------------------
__global__ void __launch_bounds__(32)
lengths_wmma_kernel(const int* __restrict__ mask, int* __restrict__ lengths) {
    const int lane = threadIdx.x;          // 0..31, single wave, EXEC all ones
    const int m    = lane & 15;            // batch row this lane feeds
    const int half = lane >> 4;            // 0: K 0..31, 1: K 32..63 of tile

    v8i bones;
#pragma unroll
    for (int i = 0; i < 8; ++i) bones[i] = 0x01010101;

    v8i c = {0, 0, 0, 0, 0, 0, 0, 0};

    const int* rowp = mask + m * TT + half * 32;

    for (int i = 0; i < TT / 64; ++i) {    // 64 WMMA iterations, K=64 each
        const int4* p = (const int4*)(rowp + i * 64);
        v8i a;
#pragma unroll
        for (int j = 0; j < 8; ++j) {
            int4 q = p[j];                 // 4 int32 mask values -> 4 u8 of (1-mask)
            a[j] = (1 - q.x) | ((1 - q.y) << 8) | ((1 - q.z) << 16) | ((1 - q.w) << 24);
        }
        // 7-arg iu8 form: (sgn_a, A, sgn_b, B, C, reuse_a, reuse_b)
        c = __builtin_amdgcn_wmma_i32_16x16x64_iu8(false, a, false, bones, c,
                                                   false, false);
    }

    // C layout (32-bit 16x16): lanes 0-15 hold M=0..7 in vgprs 0..7 (N=lane),
    // lanes 16-31 hold M=8..15. Column 0 suffices.
    if ((lane & 15) == 0) {
        int* outp = lengths + half * 8;
#pragma unroll
        for (int r = 0; r < 8; ++r) outp[r] = c[r];
    }
}

// ---------------------------------------------------------------------------
// Kernel 2: init output to -inf (covers empty batches; atomic-max identity).
// ---------------------------------------------------------------------------
__global__ void __launch_bounds__(256)
init_out_kernel(float* __restrict__ out) {
    int i = blockIdx.x * 256 + threadIdx.x;
    if (i < BB * DD) out[i] = -INFINITY;
}

// ---------------------------------------------------------------------------
// Kernel 3: streaming prefix max. 1024 blocks x 256 threads; each thread owns
// a float4 column slice and walks its 64-row time segment clamped to
// lengths[b]. 8 rows in flight per thread -> ~16 MB machine-wide in-flight
// loads across ~512 active blocks, enough to cover HBM latency at 23.3 TB/s.
// ---------------------------------------------------------------------------
__device__ inline void atomic_max_f32(float* addr, float v) {
    // Sign-split monotonic-bits trick: one hardware atomic, order-independent.
    if (v >= 0.0f) atomicMax((int*)addr, __float_as_int(v));
    else           atomicMin((unsigned int*)addr, __float_as_uint(v));
}

__device__ inline float4 max4(float4 a, float4 b) {
    float4 r;
    r.x = fmaxf(a.x, b.x); r.y = fmaxf(a.y, b.y);
    r.z = fmaxf(a.z, b.z); r.w = fmaxf(a.w, b.w);
    return r;
}

__global__ void __launch_bounds__(256)
max_reduce_kernel(const float* __restrict__ x, const int* __restrict__ lengths,
                  float* __restrict__ out) {
    const int b   = blockIdx.x >> 6;       // /NSEG
    const int s   = blockIdx.x & (NSEG - 1);
    const int len = lengths[b];

    const int t0 = s * TSEG;
    int t1 = t0 + TSEG;
    if (len < t1) t1 = len;
    if (t1 <= t0) return;                  // whole segment is padding

    const int d = threadIdx.x << 2;        // 256 threads * 4 floats = full D row
    const float* row = x + ((size_t)b * TT + (size_t)t0) * DD + d;

    const float4 ninf = {-INFINITY, -INFINITY, -INFINITY, -INFINITY};
    float4 m0 = ninf, m1 = ninf, m2 = ninf, m3 = ninf;

    const int nt = t1 - t0;
    int t = 0;
    for (; t + 8 <= nt; t += 8) {
        // gfx1250 prefetch path (global_prefetch_b8), 32 rows (128KB) ahead.
        __builtin_prefetch(row + 32 * DD, 0, 0);
        float4 v0 = *(const float4*)(row);
        float4 v1 = *(const float4*)(row + DD);
        float4 v2 = *(const float4*)(row + 2 * DD);
        float4 v3 = *(const float4*)(row + 3 * DD);
        float4 v4 = *(const float4*)(row + 4 * DD);
        float4 v5 = *(const float4*)(row + 5 * DD);
        float4 v6 = *(const float4*)(row + 6 * DD);
        float4 v7 = *(const float4*)(row + 7 * DD);
        m0 = max4(m0, max4(v0, v1));
        m1 = max4(m1, max4(v2, v3));
        m2 = max4(m2, max4(v4, v5));
        m3 = max4(m3, max4(v6, v7));
        row += 8 * DD;
    }
    for (; t < nt; ++t) {
        m0 = max4(m0, *(const float4*)row);
        row += DD;
    }
    m0 = max4(max4(m0, m1), max4(m2, m3));

    float* o = out + b * DD + d;
    atomic_max_f32(o + 0, m0.x);
    atomic_max_f32(o + 1, m0.y);
    atomic_max_f32(o + 2, m0.z);
    atomic_max_f32(o + 3, m0.w);
}

// ---------------------------------------------------------------------------
extern "C" void kernel_launch(void* const* d_in, const int* in_sizes, int n_in,
                              void* d_out, int out_size, void* d_ws, size_t ws_size,
                              hipStream_t stream) {
    (void)in_sizes; (void)n_in; (void)out_size; (void)ws_size;

    const float* x    = (const float*)d_in[0];  // [16,4096,1024] f32
    const int*   mask = (const int*)d_in[1];    // [16,4096] i32
    float*       out  = (float*)d_out;          // [16,1024] f32
    int*         lens = (int*)d_ws;             // 16 ints of scratch

    lengths_wmma_kernel<<<1, 32, 0, stream>>>(mask, lens);
    init_out_kernel<<<(BB * DD + 255) / 256, 256, 0, stream>>>(out);
    max_reduce_kernel<<<BB * NSEG, 256, 0, stream>>>(x, lens, out);
}
